// DLGN_Kernel_13030930776379
// MI455X (gfx1250) — matrix operand order repaired and checked by reading, not compiled, not run on hardware
//
#include <hip/hip_runtime.h>
#include <stdint.h>

typedef _Float16 half_t;
typedef __attribute__((ext_vector_type(16))) _Float16 v16h;
typedef __attribute__((ext_vector_type(8)))  _Float16 v8h;
typedef __attribute__((ext_vector_type(8)))  float    v8f;

#define LDP       40                 // padded LDS row stride (halfs): 80B
#define TILEB     (128 * LDP * 2)    // bytes per LDS tile buffer (10240)
#define BETA_F    4.0f
#define DIM       512
#define N_INP     4096
#define N_DATA    8192
#define N_ALL     12288
#define DEPTH     4

// ---- CDNA5 async copy: global -> LDS, 16B per lane, tracked by ASYNCcnt ----
__device__ __forceinline__ void async_ld16(uint32_t lds_byte, const void* g) {
  asm volatile("global_load_async_to_lds_b128 %0, %1, off"
               :: "v"(lds_byte), "v"(g)
               : "memory");
}
#define S_WAIT_ASYNCCNT(n) asm volatile("s_wait_asynccnt " #n ::: "memory")

// Issue one 128x32-halfs A-tile + B-tile pair into LDS (4 async instr/thread).
// Per-thread: r = tid>>2 (0..63), loff = (tid>>2)*80 + (tid&3)*16.
__device__ __forceinline__ void issue_tile(const half_t* __restrict__ A,
                                           const half_t* __restrict__ B,
                                           int i0, int j0, int k0,
                                           uint32_t ldsA, uint32_t ldsB,
                                           int r, int cc, uint32_t loff) {
  const half_t* ga0 = A + (size_t)(i0 + r)      * DIM + k0 + cc * 8;
  const half_t* ga1 = A + (size_t)(i0 + 64 + r) * DIM + k0 + cc * 8;
  const half_t* gb0 = B + (size_t)(j0 + r)      * DIM + k0 + cc * 8;
  const half_t* gb1 = B + (size_t)(j0 + 64 + r) * DIM + k0 + cc * 8;
  async_ld16(ldsA + loff,             ga0);
  async_ld16(ldsA + loff + 64 * 80,   ga1);
  async_ld16(ldsB + loff,             gb0);
  async_ld16(ldsB + loff + 64 * 80,   gb1);
}

// ---- WMMA fragment loads (wave32, v_wmma_f32_16x16x32_f16) -----------------
// A-frag (16x32 MxK): lane t -> row (t&15); k = 8*(t>>4)+{0..7}, 16+8*(t>>4)+{0..7}
__device__ __forceinline__ v16h frag_a(const half_t* base, int lane) {
  const half_t* p = base + (lane & 15) * LDP + ((lane >> 4) << 3);
  v16h f;
  ((v8h*)&f)[0] = *(const v8h*)(p);
  ((v8h*)&f)[1] = *(const v8h*)(p + 16);
  return f;
}
// B-frag (32x16 KxN) from row-major [N][K]: lane t -> col (t&15); k = 8*(t>>4)+{0..7}, 16+8*(t>>4)+{0..7}
__device__ __forceinline__ v16h frag_b(const half_t* base, int lane) {
  const half_t* p = base + (lane & 15) * LDP + ((lane >> 4) << 3);
  v16h f;
  ((v8h*)&f)[0] = *(const v8h*)(p);
  ((v8h*)&f)[1] = *(const v8h*)(p + 16);
  return f;
}

// 8 WMMAs: wave tile 32x64 (2 M-frags x 4 N-frags) from one buffered tile pair
__device__ __forceinline__ void do_wmma(const half_t* As, const half_t* Bs,
                                        int wm, int wn, int lane,
                                        v8f (&acc)[2][4]) {
  v16h af[2], bf[4];
  #pragma unroll
  for (int mt = 0; mt < 2; ++mt)
    af[mt] = frag_a(As + (wm * 32 + mt * 16) * LDP, lane);
  #pragma unroll
  for (int nt = 0; nt < 4; ++nt)
    bf[nt] = frag_b(Bs + (wn * 64 + nt * 16) * LDP, lane);
  #pragma unroll
  for (int mt = 0; mt < 2; ++mt)
    #pragma unroll
    for (int nt = 0; nt < 4; ++nt)
      acc[mt][nt] = __builtin_amdgcn_wmma_f32_16x16x32_f16(
          false, af[mt], false, bf[nt], (short)0, acc[mt][nt], false, false);
}

// ---- Phase A: activation chain  X_l = X_{l-1} @ W_l  (A@B^T, B = W^T) ------
__global__ __launch_bounds__(256) void k_chain(const half_t* __restrict__ X,
                                               const half_t* __restrict__ Wt,
                                               half_t* __restrict__ Xn,
                                               half_t* __restrict__ Act) {
  __shared__ half_t As[2][128 * LDP];
  __shared__ half_t Bs[2][128 * LDP];
  const int tid  = threadIdx.x;
  const int lane = tid & 31;
  const int w    = tid >> 5;
  const int wm   = w & 3;
  const int wn   = w >> 2;
  const int i0   = blockIdx.x * 128;
  const int j0   = blockIdx.y * 128;
  const int r    = tid >> 2;
  const int cc   = tid & 3;
  const uint32_t loff  = (uint32_t)(r * 80 + cc * 16);
  const uint32_t ldsA0 = (uint32_t)(uintptr_t)&As[0][0];
  const uint32_t ldsB0 = (uint32_t)(uintptr_t)&Bs[0][0];

  v8f acc[2][4] = {};

  issue_tile(X, Wt, i0, j0, 0, ldsA0, ldsB0, r, cc, loff);   // preload buf0

  for (int s = 0; s < 16; ++s) {
    const int buf = s & 1;
    if (s < 15) {
      issue_tile(X, Wt, i0, j0, (s + 1) * 32,
                 ldsA0 + (uint32_t)(buf ^ 1) * TILEB,
                 ldsB0 + (uint32_t)(buf ^ 1) * TILEB, r, cc, loff);
      S_WAIT_ASYNCCNT(4);
    } else {
      S_WAIT_ASYNCCNT(0);
    }
    __syncthreads();
    do_wmma(&As[buf][0], &Bs[buf][0], wm, wn, lane, acc);
    __syncthreads();
  }

  const int h = lane >> 4, lr = lane & 15;
  #pragma unroll
  for (int mt = 0; mt < 2; ++mt)
    #pragma unroll
    for (int nt = 0; nt < 4; ++nt) {
      const int col = j0 + wn * 64 + nt * 16 + lr;
      #pragma unroll
      for (int v = 0; v < 8; ++v) {
        const int row = i0 + wm * 32 + mt * 16 + h * 8 + v;
        const float x = acc[mt][nt][v];
        const size_t off = (size_t)row * DIM + col;
        Xn[off]  = (half_t)x;
        Act[off] = (half_t)(1.0f / (1.0f + __expf(-BETA_F * x)));
      }
    }
}

// ---- Phase B: fused kernel matrix + alphas contraction ---------------------
// K_tile = S0 * (S1/512) * (S2/512) * (S3/512);  out[i] += sum_j K[i,j]*alphas[j]
__global__ __launch_bounds__(256) void k_kernelmat(const half_t* __restrict__ act,
                                                   const float* __restrict__ alphas,
                                                   float* __restrict__ out) {
  __shared__ half_t As[2][128 * LDP];
  __shared__ half_t Bs[2][128 * LDP];
  const int tid  = threadIdx.x;
  const int lane = tid & 31;
  const int w    = tid >> 5;
  const int wm   = w & 3;
  const int wn   = w >> 2;
  const int i0   = blockIdx.x * 128;
  const int j0   = blockIdx.y * 128;
  const int r    = tid >> 2;
  const int cc   = tid & 3;
  const uint32_t loff  = (uint32_t)(r * 80 + cc * 16);
  const uint32_t ldsA0 = (uint32_t)(uintptr_t)&As[0][0];
  const uint32_t ldsB0 = (uint32_t)(uintptr_t)&Bs[0][0];

  v8f prod[2][4] = {};
  v8f acc[2][4]  = {};

  {
    const half_t* A0 = act;
    issue_tile(A0, A0 + (size_t)N_INP * DIM, i0, j0, 0, ldsA0, ldsB0, r, cc, loff);
  }

  // 64 flat steps: layer l = s>>4, k0 = (s&15)*32. Pipeline never drains.
  for (int s = 0; s < 64; ++s) {
    const int buf = s & 1;
    if (s < 63) {
      const int s1 = s + 1;
      const half_t* A = act + (size_t)(s1 >> 4) * N_ALL * DIM;
      issue_tile(A, A + (size_t)N_INP * DIM, i0, j0, (s1 & 15) * 32,
                 ldsA0 + (uint32_t)(buf ^ 1) * TILEB,
                 ldsB0 + (uint32_t)(buf ^ 1) * TILEB, r, cc, loff);
      S_WAIT_ASYNCCNT(4);
    } else {
      S_WAIT_ASYNCCNT(0);
    }
    __syncthreads();
    do_wmma(&As[buf][0], &Bs[buf][0], wm, wn, lane, acc);
    __syncthreads();

    if ((s & 15) == 15) {               // layer boundary: fold into product
      if (s == 15) {
        #pragma unroll
        for (int mt = 0; mt < 2; ++mt)
          #pragma unroll
          for (int nt = 0; nt < 4; ++nt)
            prod[mt][nt] = acc[mt][nt];
      } else {
        const float sc = 1.0f / 512.0f;
        #pragma unroll
        for (int mt = 0; mt < 2; ++mt)
          #pragma unroll
          for (int nt = 0; nt < 4; ++nt)
            #pragma unroll
            for (int v = 0; v < 8; ++v)
              prod[mt][nt][v] *= acc[mt][nt][v] * sc;
      }
      #pragma unroll
      for (int mt = 0; mt < 2; ++mt)
        #pragma unroll
        for (int nt = 0; nt < 4; ++nt)
          acc[mt][nt] = v8f{};
    }
  }

  // Epilogue: dot rows with alphas, reduce across the 16 column lanes.
  const int h = lane >> 4, lr = lane & 15;
  float aj[4];
  #pragma unroll
  for (int nt = 0; nt < 4; ++nt)
    aj[nt] = alphas[j0 + wn * 64 + nt * 16 + lr];

  #pragma unroll
  for (int mt = 0; mt < 2; ++mt)
    #pragma unroll
    for (int v = 0; v < 8; ++v) {
      float sacc = 0.0f;
      #pragma unroll
      for (int nt = 0; nt < 4; ++nt)
        sacc += prod[mt][nt][v] * aj[nt];
      sacc += __shfl_xor(sacc, 1);
      sacc += __shfl_xor(sacc, 2);
      sacc += __shfl_xor(sacc, 4);
      sacc += __shfl_xor(sacc, 8);
      if (lr == 0)
        atomicAdd(&out[i0 + wm * 32 + mt * 16 + h * 8 + v], sacc);
    }
}

// ---- Small prep kernels -----------------------------------------------------
__global__ void k_zero(float* __restrict__ p, int n) {
  int i = blockIdx.x * blockDim.x + threadIdx.x;
  if (i < n) p[i] = 0.0f;
}

__global__ void k_f32_to_f16(const float* __restrict__ src, half_t* __restrict__ dst, int n) {
  int i = blockIdx.x * blockDim.x + threadIdx.x;
  if (i < n) dst[i] = (half_t)src[i];
}

// gating [4][512(k)][512(n)] f32 -> gT [4][512(n)][512(k)] f16
__global__ void k_transpose_w(const float* __restrict__ g, half_t* __restrict__ gT) {
  int idx = blockIdx.x * blockDim.x + threadIdx.x;
  if (idx >= DEPTH * DIM * DIM) return;
  int l   = idx >> 18;
  int rem = idx & (DIM * DIM - 1);
  int k   = rem >> 9;
  int n   = rem & (DIM - 1);
  gT[(size_t)l * DIM * DIM + (size_t)n * DIM + k] = (half_t)g[idx];
}

// ---- Launcher ---------------------------------------------------------------
extern "C" void kernel_launch(void* const* d_in, const int* in_sizes, int n_in,
                              void* d_out, int out_size, void* d_ws, size_t ws_size,
                              hipStream_t stream) {
  (void)in_sizes; (void)n_in; (void)out_size; (void)ws_size;
  const float* inp    = (const float*)d_in[0];   // [4096, 512]
  const float* data   = (const float*)d_in[1];   // [8192, 512]
  const float* gating = (const float*)d_in[2];   // [4, 512, 512]
  const float* alphas = (const float*)d_in[3];   // [8192]
  float* out = (float*)d_out;                    // [4096]

  half_t* gT  = (half_t*)d_ws;                         // 2 MB
  half_t* X0  = gT + (size_t)DEPTH * DIM * DIM;        // 12 MB ping
  half_t* X1  = X0 + (size_t)N_ALL * DIM;              // 12 MB pong
  half_t* Act = X1 + (size_t)N_ALL * DIM;              // 48 MB

  k_zero<<<dim3((N_INP + 255) / 256), dim3(256), 0, stream>>>(out, N_INP);

  k_f32_to_f16<<<dim3((N_INP * DIM) / 256), dim3(256), 0, stream>>>(inp, X0, N_INP * DIM);
  k_f32_to_f16<<<dim3((N_DATA * DIM) / 256), dim3(256), 0, stream>>>(
      data, X0 + (size_t)N_INP * DIM, N_DATA * DIM);
  k_transpose_w<<<dim3((DEPTH * DIM * DIM) / 256), dim3(256), 0, stream>>>(gating, gT);

  half_t* cur = X0;
  half_t* nxt = X1;
  for (int l = 0; l < DEPTH; ++l) {
    k_chain<<<dim3(N_ALL / 128, DIM / 128), dim3(256), 0, stream>>>(
        cur, gT + (size_t)l * DIM * DIM, nxt, Act + (size_t)l * N_ALL * DIM);
    half_t* t = cur; cur = nxt; nxt = t;
  }

  k_kernelmat<<<dim3(N_INP / 128, N_DATA / 128), dim3(256), 0, stream>>>(Act, alphas, out);
}
